// BrainwaveProcessor_61392262529462
// MI455X (gfx1250) — matrix operand order repaired and checked by reading, not compiled, not run on hardware
//
#include <hip/hip_runtime.h>
#include <math.h>

#define BATCH 16
#define SEQ   2048
#define DIM   512
#define NGRP  8
#define GSZ   64             // DIM / NGRP
#define KW    5
#define KDIM  (GSZ * KW)     // 320 reduction length (in-ch x tap)
#define KSTEP 32
#define NKS   (KDIM / KSTEP) // 10 WMMA steps
#define NMT   4              // 64 out-ch per group / 16 rows per WMMA tile

typedef __attribute__((ext_vector_type(16))) _Float16 v16h;
typedef __attribute__((ext_vector_type(8)))  float    v8f;

// packed-W element count: groups * mtiles * ksteps * 32 lanes * 16 halfs
#define WPACK_ELEMS (NGRP * NMT * NKS * 32 * 16)   // 512*320 = 163840

// ---------------------------------------------------------------------------
// Kernel 0: fnorm = freq_filter / max(||freq_filter||2, 1e-12)
// ---------------------------------------------------------------------------
__global__ void __launch_bounds__(DIM) k_fnorm(const float* __restrict__ ff,
                                               float* __restrict__ fnorm) {
  __shared__ float red[DIM];
  const int t = threadIdx.x;
  const float v = ff[t];
  red[t] = v * v;
  __syncthreads();
  for (int off = DIM / 2; off > 0; off >>= 1) {
    if (t < off) red[t] += red[t + off];
    __syncthreads();
  }
  const float nrm = fmaxf(sqrtf(red[0]), 1e-12f);
  fnorm[t] = v / nrm;
}

// ---------------------------------------------------------------------------
// Kernel 0b: pack conv_w (f32, [D][GSZ][KW]) into f16 laid out in contiguous-k WMMA
// A-fragment order: [g][mtile][kstep][lane][16 halfs]. Each lane's fragment
// slice becomes one contiguous 32-byte global_load_b128 pair in k_conv.
// ---------------------------------------------------------------------------
__global__ void __launch_bounds__(256) k_packw(const float* __restrict__ w,
                                               _Float16* __restrict__ wpack) {
  int idx = blockIdx.x * 256 + threadIdx.x;
  if (idx >= WPACK_ELEMS) return;
  const int j     = idx & 15;
  const int lane  = (idx >> 4) & 31;
  const int t     = (idx >> 9) % NKS;
  const int mtile = (idx / (16 * 32 * NKS)) % NMT;
  const int g     = idx / (16 * 32 * NKS * NMT);

  const int abase = (lane & 16) ? 16 : 0;  // K-base for upper half-wave
  const int ka = t * KSTEP + abase + j;
  const int k = ka >> 6, i = ka & 63;      // reduction r = k*64 + i
  const int d = g * GSZ + mtile * 16 + (lane & 15);
  wpack[idx] = (_Float16)w[(size_t)d * KDIM + i * KW + k];
}

// ---------------------------------------------------------------------------
// Kernel 1: grouped conv1d (K=5, groups=8) as WMMA GEMM + tanh + residual.
// Grid: (SEQ/16, NGRP, BATCH), block 128 (4 waves, one 16x16 out tile each).
// out[b,s,d] = tanh(conv + bias) + 0.1*filtered   (written to d_out)
// ---------------------------------------------------------------------------
__global__ void __launch_bounds__(128) k_conv(const float* __restrict__ x,
                                              const float* __restrict__ fnorm,
                                              const _Float16* __restrict__ wpack,
                                              const float* __restrict__ bias,
                                              float* __restrict__ out) {
  // filtered input patch: 16 positions + K-1 halo = 20 rows x 64 in-ch, fp16
  __shared__ __attribute__((aligned(32))) _Float16 sX[16 + KW - 1][GSZ];

  const int st = blockIdx.x;            // seq tile index
  const int g  = blockIdx.y;            // group
  const int b  = blockIdx.z;            // batch
  const int s0 = st * 16;
  const int gbase = g * GSZ;

  // cooperative fill: filtered = clip(x * fnorm, +-10); zero pad at seq edges
  for (int e = threadIdx.x; e < 20 * GSZ; e += 128) {
    const int p = e >> 6, i = e & 63;
    const int s = s0 + p - (KW / 2);
    float v = 0.f;
    if (s >= 0 && s < SEQ) {
      v = x[((size_t)b * SEQ + s) * DIM + gbase + i] * fnorm[gbase + i];
      v = fminf(fmaxf(v, -10.f), 10.f);
    }
    sX[p][i] = (_Float16)v;
  }
  __syncthreads();

  const int lane  = threadIdx.x & 31;
  const int mtile = threadIdx.x >> 5;   // 0..3 -> 16 out channels each
  const int n     = lane & 15;          // B/C: N index; A: M index
  const int abase = (lane & 16) ? 8 : 0;
  const int bbase = (lane & 16) ? 16 : 0;

  // per-lane base into packed A fragments (contiguous 16 halfs per K-step)
  const _Float16* wlane =
      wpack + ((size_t)((g * NMT + mtile) * NKS) * 32 + lane) * 16;

  v8f acc = {};
#pragma unroll
  for (int t = 0; t < NKS; ++t) {
    // A: one 32-byte contiguous load from the packed weights (L2-resident)
    const v16h a = *(const v16h*)(wlane + (size_t)t * (32 * 16));
    // B: fragment's 16 K-elements never cross a 64-elem tap boundary, so it
    // is a single aligned 32-byte LDS load: row = n + tap, cols i2..i2+15
    const int kb0 = t * KSTEP + bbase;
    const int k2 = kb0 >> 6, i2 = kb0 & 63;
    const v16h bm = *(const v16h*)&sX[n + k2][i2];
    acc = __builtin_amdgcn_wmma_f32_16x16x32_f16(
        /*neg_a=*/false, a, /*neg_b=*/false, bm,
        /*c_mod=*/(short)0, acc, /*reuse_a=*/false, /*reuse_b=*/false);
  }

  // C/D layout: VGPR v -> M = v + 8*(lane>=16), N = lane&15
#pragma unroll
  for (int v = 0; v < 8; ++v) {
    const int o = mtile * 16 + v + abase;       // out channel within group
    const int d = gbase + o;
    const int s = s0 + n;
    const float val =
        tanhf(acc[v] + bias[d]) + 0.1f * (float)sX[n + (KW / 2)][o];
    out[((size_t)b * SEQ + s) * DIM + d] = val;
  }
}

// ---------------------------------------------------------------------------
// Kernel 2: rfft -> 1/f^0.5 scaling -> irfft as a full complex Stockham FFT
// (length 2048) in LDS with symmetric-bin scaling. In place.
// One block (256 threads) per (b, d) column.
// ---------------------------------------------------------------------------
__device__ __forceinline__ int stockham(float (*re)[SEQ], float (*im)[SEQ],
                                        int cur, float sign, int tid) {
  for (int lgs = 0, ncur = SEQ; ncur > 1; ncur >>= 1, ++lgs) {
    const int m = ncur >> 1;
    const int stride = 1 << lgs;
    const int nxt = cur ^ 1;
    for (int tix = tid; tix < SEQ / 2; tix += 256) {
      const int p = tix >> lgs;
      const int q = tix & (stride - 1);
      const float ang = sign * 6.28318530717958647692f * (float)p / (float)ncur;
      const float c = __cosf(ang), sn = __sinf(ang);
      const int i0 = q + stride * p;
      const int i1 = q + stride * (p + m);
      const int o0 = q + stride * (2 * p);
      const int o1 = o0 + stride;
      const float ar = re[cur][i0], ai = im[cur][i0];
      const float br = re[cur][i1], bi = im[cur][i1];
      re[nxt][o0] = ar + br;
      im[nxt][o0] = ai + bi;
      const float dr = ar - br, di = ai - bi;
      re[nxt][o1] = dr * c - di * sn;
      im[nxt][o1] = dr * sn + di * c;
    }
    __syncthreads();
    cur ^= 1;
  }
  return cur;
}

__global__ void __launch_bounds__(256) k_fft(float* __restrict__ data) {
  __shared__ float sre[2][SEQ];
  __shared__ float sim[2][SEQ];
  const int bd = blockIdx.x;
  const int b = bd / DIM, d = bd % DIM;
  const int tid = threadIdx.x;
  float* col = data + (size_t)b * SEQ * DIM + d;

  for (int s = tid; s < SEQ; s += 256) {
    sre[0][s] = col[(size_t)s * DIM];
    sim[0][s] = 0.f;
  }
  __syncthreads();

  int cur = stockham(sre, sim, 0, -1.f, tid);   // forward DFT

  // 1 / clip((|f|+eps)^0.5, eps, 1e6), mirrored onto negative bins
  for (int kf = tid; kf < SEQ; kf += 256) {
    const int kk = (kf <= SEQ / 2) ? kf : (SEQ - kf);
    const float fr = (float)kk * (1.0f / SEQ);
    float den = sqrtf(fr + 1e-8f);
    den = fminf(fmaxf(den, 1e-8f), 1e6f);
    const float sc = 1.0f / den;
    sre[cur][kf] *= sc;
    sim[cur][kf] *= sc;
  }
  __syncthreads();

  cur = stockham(sre, sim, cur, 1.f, tid);      // inverse DFT (x N)

  const float invn = 1.0f / (float)SEQ;
  for (int s = tid; s < SEQ; s += 256) {
    float v = sre[cur][s] * invn;
    v = fminf(fmaxf(v, -10.f), 10.f);
    col[(size_t)s * DIM] = v;
  }
}

// ---------------------------------------------------------------------------
// Kernel 3: LayerNorm over D (biased variance), in place. Block per (b,s).
// ---------------------------------------------------------------------------
__global__ void __launch_bounds__(256) k_ln(float* __restrict__ data,
                                            const float* __restrict__ gamma,
                                            const float* __restrict__ beta) {
  __shared__ float s1[256], s2[256];
  const int row = blockIdx.x;                 // b*SEQ + s
  const int t = threadIdx.x;
  float* r = data + (size_t)row * DIM;
  const float a0 = r[t], a1 = r[t + 256];
  s1[t] = a0 + a1;
  s2[t] = a0 * a0 + a1 * a1;
  __syncthreads();
  for (int off = 128; off > 0; off >>= 1) {
    if (t < off) { s1[t] += s1[t + off]; s2[t] += s2[t + off]; }
    __syncthreads();
  }
  const float mu = s1[0] * (1.0f / DIM);
  const float var = s2[0] * (1.0f / DIM) - mu * mu;
  const float rs = rsqrtf(var + 1e-5f);
  r[t]       = (a0 - mu) * rs * gamma[t] + beta[t];
  r[t + 256] = (a1 - mu) * rs * gamma[t + 256] + beta[t + 256];
}

// ---------------------------------------------------------------------------
// Kernel 4: temporal EMA scan (a = 1/16), in place. Thread per (b,d) column,
// coalesced across d; prefetch next row ahead of the dependent chain.
// ---------------------------------------------------------------------------
__global__ void __launch_bounds__(DIM) k_ema(float* __restrict__ data) {
  const int b = blockIdx.x;
  const int d = threadIdx.x;
  float* col = data + (size_t)b * SEQ * DIM + d;
  const float a = 1.0f / 16.0f;
  float m = col[0];
  for (int t = 1; t < SEQ; ++t) {
    if (t + 1 < SEQ) __builtin_prefetch(col + (size_t)(t + 1) * DIM, 0, 1);
    const float v = col[(size_t)t * DIM];
    m = a * v + (1.0f - a) * m;
    col[(size_t)t * DIM] = m;
  }
}

// ---------------------------------------------------------------------------
extern "C" void kernel_launch(void* const* d_in, const int* in_sizes, int n_in,
                              void* d_out, int out_size, void* d_ws,
                              size_t ws_size, hipStream_t stream) {
  (void)in_sizes; (void)n_in; (void)out_size; (void)ws_size;
  const float* x     = (const float*)d_in[0];
  const float* ff    = (const float*)d_in[1];
  const float* convw = (const float*)d_in[2];
  const float* convb = (const float*)d_in[3];
  const float* gamma = (const float*)d_in[4];
  const float* beta  = (const float*)d_in[5];
  float* out = (float*)d_out;

  // workspace layout: [0, 2KB) fnorm (DIM floats), then packed f16 weights
  float*    fnorm = (float*)d_ws;
  _Float16* wpack = (_Float16*)((char*)d_ws + DIM * sizeof(float));

  k_fnorm<<<1, DIM, 0, stream>>>(ff, fnorm);
  k_packw<<<(WPACK_ELEMS + 255) / 256, 256, 0, stream>>>(convw, wpack);

  dim3 cgrid(SEQ / 16, NGRP, BATCH);
  k_conv<<<cgrid, 128, 0, stream>>>(x, fnorm, wpack, convb, out);

  k_fft<<<BATCH * DIM, 256, 0, stream>>>(out);

  k_ln<<<BATCH * SEQ, 256, 0, stream>>>(out, gamma, beta);

  k_ema<<<BATCH, DIM, 0, stream>>>(out);
}
